// BlockdiagButterflyLinear_47304769798209
// MI455X (gfx1250) — matrix operand order repaired and checked by reading, not compiled, not run on hardware
//
#include <hip/hip_runtime.h>
#include <hip/hip_bf16.h>
#include <stddef.h>

// ---------------------------------------------------------------------------
// Monarch block-diagonal butterfly:  out = (blockdiag2(blockdiag1(x)^T)) + bias
//   x:  (16384, 4096) fp32
//   w1: (4, 192, 1024) fp32   stage-1 blocks (q=l*48+b1 output features, K=1024)
//   w2: (4, 1024, 192) fp32   stage-2 blocks (s output features, K=192)
// Strategy: bf16 WMMA (v_wmma_f32_16x16x32_bf16) with f32 accum.
//   A = weights (M = out features), B = batch columns, so C-fragment lanes hold
//   8 consecutive features for one batch row -> packed vector stores.
//   Block transpose fused into stage-1 store index: q=l*48+b1 -> l*192+k*48+b1.
//   Intermediate kept in bf16 in d_ws to halve its HBM traffic.
// ---------------------------------------------------------------------------

typedef __bf16 bf16_t;
typedef __attribute__((ext_vector_type(16))) __bf16 v16bf;
typedef __attribute__((ext_vector_type(8)))  __bf16 v8bf;
typedef __attribute__((ext_vector_type(8)))  float  v8f;
typedef __attribute__((ext_vector_type(4)))  float  v4f;

#define IN_F   4096
#define OUT_F  4096
#define NB     4
#define PP     1024   // stage-1 K per block
#define QQ     192    // stage-1 out features per block
#define SS     1024   // stage-2 out features per block
#define RR     192    // stage-2 K per block
#define BB1    48
#define MIDW   768    // intermediate row width (l*192)

static __device__ __forceinline__ v16bf cat8(v8bf lo, v8bf hi) {
    v16bf r;
#pragma unroll
    for (int i = 0; i < 8; ++i) { r[i] = lo[i]; r[i + 8] = hi[i]; }
    return r;
}

// A fragment (16x32 bf16) from row-major (rows=M features, K contiguous).
// Lanes 0-15: M=lane, K=[k..k+16) contiguous.
// Lanes 16-31: M=lane-16, K=[k+16..k+32).
static __device__ __forceinline__ v16bf load_a(const bf16_t* __restrict__ p) {
    return cat8(*(const v8bf*)p, *(const v8bf*)(p + 8));
}

// ---------------------------------------------------------------------------
// weight fp32 -> bf16 conversion (one-shot, tiny traffic)
// ---------------------------------------------------------------------------
__global__ void __launch_bounds__(256)
cvt_weights_kernel(const float* __restrict__ w1, const float* __restrict__ w2,
                   bf16_t* __restrict__ w1b, bf16_t* __restrict__ w2b, int n) {
    int i = blockIdx.x * 256 + threadIdx.x;
    if (i < n) {
        w1b[i] = (bf16_t)w1[i];
        w2b[i] = (bf16_t)w2[i];
    }
}

// ---------------------------------------------------------------------------
// Stage 1: per k-block GEMM  mid[b, l*192 + k*48 + b1] = x[b, k*1024+..] @ w1[k]^T
// Block = 256 threads = 8 waves; each wave: 16 batch rows x all 192 features.
// Grid: (batch/128, 4)
// ---------------------------------------------------------------------------
__global__ void __launch_bounds__(256)
stage1_kernel(const float* __restrict__ x, const bf16_t* __restrict__ w1b,
              bf16_t* __restrict__ mid) {
    const int lane = threadIdx.x & 31;
    const int wave = threadIdx.x >> 5;
    const int k    = blockIdx.y;
    const int bbase = blockIdx.x * 128 + wave * 16;
    const int n  = lane & 15;   // batch row within tile / weight row within tile
    const int hi = lane >> 4;

    // B-fragment source: x row (bbase+n), k-th 1024-wide block, fp32
    const float* xp = x + (size_t)(bbase + n) * IN_F + k * PP + hi * 16;
    // A-fragment source: w1[k] row-major (192 x 1024) bf16
    const bf16_t* wp = w1b + (size_t)k * QQ * PP + (size_t)n * PP + hi * 16;

    v8f acc[12];
#pragma unroll
    for (int t = 0; t < 12; ++t) acc[t] = (v8f){};

    for (int kk = 0; kk < PP; kk += 32) {
        // B fragment: 16 contiguous fp32 -> bf16 (lane n = batch col)
        v4f f0 = *(const v4f*)(xp + kk + 0);
        v4f f1 = *(const v4f*)(xp + kk + 4);
        v4f f2 = *(const v4f*)(xp + kk + 8);
        v4f f3 = *(const v4f*)(xp + kk + 12);
        v16bf bfrag;
#pragma unroll
        for (int i = 0; i < 4; ++i) {
            bfrag[i]      = (bf16_t)f0[i];
            bfrag[i + 4]  = (bf16_t)f1[i];
            bfrag[i + 8]  = (bf16_t)f2[i];
            bfrag[i + 12] = (bf16_t)f3[i];
        }
#pragma unroll
        for (int t = 0; t < 12; ++t) {
            const bf16_t* ap = wp + (size_t)t * 16 * PP + kk;
            v16bf afrag = load_a(ap);
            acc[t] = __builtin_amdgcn_wmma_f32_16x16x32_bf16(
                false, afrag, false, bfrag, (short)0, acc[t], false, false);
        }
    }

    // Epilogue: fused block-transpose store, bf16, 8 features per lane (b128)
    const int brow = bbase + n;
#pragma unroll
    for (int t = 0; t < 12; ++t) {
        int q0  = t * 16 + hi * 8;      // feature index within 192
        int l   = q0 / BB1;
        int b1i = q0 - l * BB1;
        v8bf o;
#pragma unroll
        for (int j = 0; j < 8; ++j) o[j] = (bf16_t)acc[t][j];
        *(v8bf*)(mid + (size_t)brow * MIDW + l * QQ + k * BB1 + b1i) = o;
    }
}

// ---------------------------------------------------------------------------
// Stage 2: per l-block GEMM  out[b, l*1024 + s] = mid[b, l*192+..] @ w2[l]^T + bias
// Block = 256 threads = 8 waves: 2 batch sub-tiles x 4 feature groups (256 each).
// Grid: (batch/32, 4)
// ---------------------------------------------------------------------------
__global__ void __launch_bounds__(256)
stage2_kernel(const bf16_t* __restrict__ mid, const bf16_t* __restrict__ w2b,
              const float* __restrict__ bias, float* __restrict__ out) {
    const int lane  = threadIdx.x & 31;
    const int wave  = threadIdx.x >> 5;
    const int l     = blockIdx.y;
    const int bbase = blockIdx.x * 32 + (wave & 1) * 16;
    const int fbase = (wave >> 1) * 256;
    const int n  = lane & 15;
    const int hi = lane >> 4;

    // B-fragment source: mid row (bbase+n), l-th 192-wide segment, bf16
    const bf16_t* bp = mid + (size_t)(bbase + n) * MIDW + l * RR + hi * 16;
    // A-fragment source: w2[l] row-major (1024 x 192) bf16
    const bf16_t* ap0 = w2b + (size_t)l * SS * RR + (size_t)(fbase + n) * RR + hi * 16;

    v8f acc[16];
#pragma unroll
    for (int t = 0; t < 16; ++t) acc[t] = (v8f){};

    for (int kk = 0; kk < RR; kk += 32) {
        v16bf bfrag = cat8(*(const v8bf*)(bp + kk), *(const v8bf*)(bp + kk + 8));
#pragma unroll
        for (int t = 0; t < 16; ++t) {
            const bf16_t* ap = ap0 + (size_t)t * 16 * RR + kk;
            v16bf afrag = load_a(ap);
            acc[t] = __builtin_amdgcn_wmma_f32_16x16x32_bf16(
                false, afrag, false, bfrag, (short)0, acc[t], false, false);
        }
    }

    // Epilogue: bias add, fp32 output, 8 consecutive features per lane
    const int brow = bbase + n;
#pragma unroll
    for (int t = 0; t < 16; ++t) {
        int s0 = fbase + t * 16 + hi * 8;
        const float* bptr = bias + l * SS + s0;
        v4f c0 = *(const v4f*)bptr;
        v4f c1 = *(const v4f*)(bptr + 4);
        v4f o0, o1;
#pragma unroll
        for (int j = 0; j < 4; ++j) {
            o0[j] = acc[t][j]     + c0[j];
            o1[j] = acc[t][j + 4] + c1[j];
        }
        float* op = out + (size_t)brow * OUT_F + l * SS + s0;
        *(v4f*)op       = o0;
        *(v4f*)(op + 4) = o1;
    }
}

// ---------------------------------------------------------------------------
extern "C" void kernel_launch(void* const* d_in, const int* in_sizes, int n_in,
                              void* d_out, int out_size, void* d_ws, size_t ws_size,
                              hipStream_t stream) {
    const float* x    = (const float*)d_in[0];
    const float* w1   = (const float*)d_in[1];
    const float* w2   = (const float*)d_in[2];
    const float* bias = (const float*)d_in[3];
    float* out = (float*)d_out;

    const int batch = in_sizes[0] / IN_F;          // 16384
    const int nw    = NB * QQ * PP;                // 786432 (== NB*SS*RR)

    // workspace layout: [w1 bf16 | w2 bf16 | mid bf16]
    bf16_t* w1b = (bf16_t*)d_ws;
    bf16_t* w2b = w1b + nw;
    bf16_t* mid = w2b + nw;

    cvt_weights_kernel<<<(nw + 255) / 256, 256, 0, stream>>>(w1, w2, w1b, w2b, nw);

    dim3 g1(batch / 128, NB);
    stage1_kernel<<<g1, 256, 0, stream>>>(x, w1b, mid);

    dim3 g2(batch / 32, NB);
    stage2_kernel<<<g2, 256, 0, stream>>>(mid, w2b, bias, out);
}
